// CFChurn11_89859305767617
// MI455X (gfx1250) — matrix operand order repaired and checked by reading, not compiled, not run on hardware
//
#include <hip/hip_runtime.h>

typedef __attribute__((ext_vector_type(16))) _Float16 v16h;
typedef __attribute__((ext_vector_type(8)))  float    v8f;

template<typename T> struct V8T;
template<> struct V8T<float>    { typedef float    type __attribute__((ext_vector_type(8))); };
template<> struct V8T<_Float16> { typedef _Float16 type __attribute__((ext_vector_type(8))); };

// ======================================================================
// Weight pre-pack: W (K x M, row-major f32, leading dim ldw) ->
// Wp[chunk][col][32] f16, zero-padded so K is a multiple of 32.
// In the GEMM, lane (lh,lm) loads B half h (k = chunk*32 + lh*8 + h + (h>=8?8:0)) from slot lh*16 + h,
// i.e. 16 *contiguous* halves = one 32B vector load per sub-tile.
// ======================================================================
__global__ void k_packW(const float* __restrict__ W, int ldw, int K, int M,
                        _Float16* __restrict__ Wp, long total)
{
  long idx = (long)blockIdx.x * blockDim.x + threadIdx.x;
  if (idx >= total) return;
  int  kk = (int)(idx & 31);
  long t  = idx >> 5;
  int  c  = (int)(t % M);
  int  ch = (int)(t / M);
  int  k  = ch * 32 + ((kk >> 3) & 1) * 16 + (kk >> 4) * 8 + (kk & 7);
  Wp[idx] = (k < K) ? (_Float16)W[(long)k * ldw + c] : (_Float16)0.f;
}

// ======================================================================
// WMMA GEMM: Out = epi( act( (X [+Xadd])*rowScale @ W + bias + addPre[preIdx] )
//                       + addPost )   [optional scatter-atomic output]
// One 16x64 output stripe per wave32 (4 x v_wmma_f32_16x16x32_f16 per
// 32-wide K chunk; A packed once per chunk and reused by 4 sub-tiles).
// A layout (16-bit 16x32): lane {lh=lane>>4, lm=lane&15}; half h ->
//   k = lh*8 + h + (h>=8?8:0)  => two contiguous 8-elem runs per lane.
// B comes from the packed f16 buffer (single 32B load per sub-tile).
// C/D f32: vgpr i -> row m = lh*8 + i, col n = lm.
// ======================================================================
template<typename TX>
__global__ __launch_bounds__(256) void wmma_gemm_kernel(
    const TX* __restrict__ X, int ldx,
    const TX* __restrict__ Xadd,
    const float* __restrict__ rowScale,
    const _Float16* __restrict__ Wp,
    const float* __restrict__ bias,
    const float* __restrict__ addPre, const int* __restrict__ preIdx,
    const float* __restrict__ addPost,
    float* __restrict__ Out, int ldo,
    const int* __restrict__ scatIdx,
    int Nrows, int K, int M, int act, int tilesMG)
{
  typedef typename V8T<TX>::type V8;
  long wave = (long)blockIdx.x * (blockDim.x >> 5) + (threadIdx.x >> 5);
  int  lane = threadIdx.x & 31;
  long tileR = wave / tilesMG;
  int  tileG = (int)(wave % tilesMG);
  if ((long)tileR * 16 >= (long)Nrows) return;   // wave-uniform exit
  int lm = lane & 15, lh = lane >> 4;

  int r  = (int)tileR * 16 + lm;
  int rc = r < Nrows ? r : Nrows - 1;            // clamp, mask at store
  int cb = tileG * 64;

  int cc[4];                                     // clamped B columns
#pragma unroll
  for (int j = 0; j < 4; ++j) {
    int c = cb + j * 16 + lm;
    cc[j] = c < M ? c : M - 1;
  }

  float rs = rowScale ? rowScale[rc] : 1.0f;
  const TX* xrow  = X + (long)rc * ldx;
  const TX* xarow = Xadd ? (Xadd + (long)rc * ldx) : (const TX*)0;

  v8f acc0 = {0.f,0.f,0.f,0.f,0.f,0.f,0.f,0.f};
  v8f acc1 = acc0, acc2 = acc0, acc3 = acc0;

  const int kFull = K & ~31;
  int ch = 0;
  for (int k0 = 0; k0 < kFull; k0 += 32, ++ch) {
    // ---- pack A: two contiguous 8-wide vector loads per lane ----
    int abase = k0 + lh * 8;
    V8 lo = *(const V8*)(xrow + abase);
    V8 hi = *(const V8*)(xrow + abase + 16);
    if (xarow) {                                 // uniform branch
      V8 lo2 = *(const V8*)(xarow + abase);
      V8 hi2 = *(const V8*)(xarow + abase + 16);
      lo += lo2; hi += hi2;
    }
    v16h a;
#pragma unroll
    for (int h = 0; h < 8; ++h) a[h]     = (_Float16)((float)lo[h] * rs);
#pragma unroll
    for (int h = 0; h < 8; ++h) a[8 + h] = (_Float16)((float)hi[h] * rs);

    // ---- 4 column sub-tiles share this A; B = one 32B packed load ----
#pragma unroll
    for (int j = 0; j < 4; ++j) {
      v16h b = *(const v16h*)(Wp + (((long)ch * M + cc[j]) << 5) + (lh << 4));
      v8f* accp = (j == 0) ? &acc0 : (j == 1) ? &acc1 : (j == 2) ? &acc2 : &acc3;
      *accp = __builtin_amdgcn_wmma_f32_16x16x32_f16(false, a, false, b,
                                                     (short)0, *accp, false, false);
    }
  }

  if (kFull < K) {                               // A tail: clamp + select
    v16h a;
    int abase = kFull + lh * 8;
#pragma unroll
    for (int h = 0; h < 16; ++h) {
      int k  = abase + h + ((h >= 8) ? 8 : 0);
      int kc = k < K ? k : K - 1;
      float v = (float)xrow[kc];
      if (xarow) v += (float)xarow[kc];          // uniform branch
      a[h] = (_Float16)((k < K) ? v * rs : 0.f);
    }
#pragma unroll
    for (int j = 0; j < 4; ++j) {
      v16h b = *(const v16h*)(Wp + (((long)ch * M + cc[j]) << 5) + (lh << 4));
      v8f* accp = (j == 0) ? &acc0 : (j == 1) ? &acc1 : (j == 2) ? &acc2 : &acc3;
      *accp = __builtin_amdgcn_wmma_f32_16x16x32_f16(false, a, false, b,
                                                     (short)0, *accp, false, false);
    }
  }

  // ---- epilogue ----
#pragma unroll
  for (int j = 0; j < 4; ++j) {
    v8f accv = (j == 0) ? acc0 : (j == 1) ? acc1 : (j == 2) ? acc2 : acc3;
#pragma unroll
    for (int i = 0; i < 8; ++i) {
      int ro = (int)tileR * 16 + lh * 8 + i;
      int co = cb + j * 16 + lm;
      if (ro < Nrows && co < M) {
        float v = accv[i];
        if (bias)   v += bias[co];
        if (addPre) { int pr = preIdx ? preIdx[ro] : ro; v += addPre[(long)pr * M + co]; }
        if (act == 1) v = fmaxf(v, 0.f);
        if (addPost) v += addPost[(long)ro * ldo + co];
        if (scatIdx) atomicAdd(&Out[(long)scatIdx[ro] * ldo + co], v); // fused seg-sum
        else         Out[(long)ro * ldo + co] = v;
      }
    }
  }
}

// ======================================================================
// Small VALU kernels
// ======================================================================
__global__ void k_embed(const float* __restrict__ xd, const float* __restrict__ cont,
                        const float* __restrict__ Wc, const float* __restrict__ bc,
                        float* __restrict__ x, int n)
{
  int i = blockIdx.x * blockDim.x + threadIdx.x;
  if (i >= n) return;
  float* xr = x + (long)i * 120;
  const float* dr = xd + (long)i * 38;
  for (int c = 0; c < 32; ++c) xr[c] = dr[c];
  const float* cr = cont + (long)i * 48;
  for (int w = 0; w < 3; ++w)
    for (int c = 0; c < 8; ++c) {
      float acc = bc[c];
      for (int f = 0; f < 16; ++f) acc += cr[w*16 + f] * Wc[f*8 + c];
      xr[32 + w*8 + c] = fmaxf(acc, 0.f);
    }
}

__global__ void k_count(const int* __restrict__ dst, float* __restrict__ cnt, int E)
{
  int i = blockIdx.x * blockDim.x + threadIdx.x;
  if (i < E) atomicAdd(&cnt[dst[i]], 1.f);
}

__global__ void k_deg(const float* __restrict__ cnt, float* __restrict__ deg,
                      float* __restrict__ inv_deg, float* __restrict__ inv_degm, int n)
{
  int i = blockIdx.x * blockDim.x + threadIdx.x;
  if (i >= n) return;
  float c = cnt[i];
  deg[i] = c + 1.f;
  inv_deg[i]  = 1.f / (c + 1.f);
  inv_degm[i] = 1.f / fmaxf(c, 1.f);
}

__global__ void k_gcn_scatter(const int* __restrict__ src, const int* __restrict__ dst,
                              const float* __restrict__ h, const float* __restrict__ deg,
                              float* __restrict__ agg, int E)
{
  long idx = (long)blockIdx.x * blockDim.x + threadIdx.x;
  if (idx >= (long)E * 64) return;
  int e = (int)(idx >> 6), c = (int)(idx & 63);
  int s = src[e], d = dst[e];
  float norm = rsqrtf(deg[s] * deg[d]);
  atomicAdd(&agg[(long)d * 64 + c], h[(long)s * 64 + c] * norm);
}

__global__ void k_gcn_combine(const float* __restrict__ agg, const float* __restrict__ h,
                              const float* __restrict__ inv_deg, const float* __restrict__ b,
                              float* __restrict__ out, int n)
{
  long idx = (long)blockIdx.x * blockDim.x + threadIdx.x;
  if (idx >= (long)n * 64) return;
  int node = (int)(idx >> 6), c = (int)(idx & 63);
  out[idx] = fmaxf(agg[idx] + h[idx] * inv_deg[node] + b[c], 0.f);
}

__global__ void k_xsum(const float* __restrict__ g0, const float* __restrict__ g1,
                       float* __restrict__ x, int n)
{
  long idx = (long)blockIdx.x * blockDim.x + threadIdx.x;
  if (idx >= (long)n * 64) return;
  int node = (int)(idx >> 6), c = (int)(idx & 63);
  x[(long)node * 120 + 56 + c] = g0[idx] + g1[idx];
}

__global__ void k_cross(const float* __restrict__ x, const float* __restrict__ xin,
                        const float* __restrict__ w, const float* __restrict__ b,
                        float* __restrict__ xout, int n)
{
  int node = blockIdx.x, tid = threadIdx.x;     // blockDim = 128
  __shared__ float red[128];
  float v = (tid < 120) ? xin[(long)node * 120 + tid] * w[tid] : 0.f;
  red[tid] = v; __syncthreads();
  for (int s = 64; s > 0; s >>= 1) { if (tid < s) red[tid] += red[tid + s]; __syncthreads(); }
  float sv = red[0];
  if (tid < 120) {
    long i = (long)node * 120 + tid;
    xout[i] = x[i] * sv + b[tid] + xin[i];
  }
}

__global__ void k_edge_e(const int* __restrict__ src, const int* __restrict__ dst,
                         const float* __restrict__ A1, const float* __restrict__ A2,
                         const float* __restrict__ ea, const float* __restrict__ Wb,
                         const float* __restrict__ be, _Float16* __restrict__ e, int E)
{
  long idx = (long)blockIdx.x * blockDim.x + threadIdx.x;
  if (idx >= (long)E * 64) return;
  int ed = (int)(idx >> 6), c = (int)(idx & 63);
  int s = src[ed], d = dst[ed];
  float v = A1[(long)s * 64 + c] + A2[(long)d * 64 + c] + be[c];
  const float* er = ea + (long)ed * 8;
#pragma unroll
  for (int f = 0; f < 8; ++f) v += er[f] * Wb[f * 64 + c];
  e[idx] = (_Float16)fmaxf(v, 0.f);
}

__global__ void k_si_combine(const float* __restrict__ agg, const float* __restrict__ inv_degm,
                             const float* __restrict__ xrt, const float* __restrict__ b,
                             float* __restrict__ out, int n)
{
  long idx = (long)blockIdx.x * blockDim.x + threadIdx.x;
  if (idx >= (long)n * 64) return;
  int node = (int)(idx >> 6), c = (int)(idx & 63);
  out[idx] = fmaxf(agg[idx] * inv_degm[node] + xrt[idx] + b[c], 0.f);
}

__global__ void k_gate_scatter(const int* __restrict__ src, const int* __restrict__ dst,
                               const float* __restrict__ cd, const float* __restrict__ x,
                               float* __restrict__ agg, int E)
{
  long idx = (long)blockIdx.x * blockDim.x + threadIdx.x;
  if (idx >= (long)E * 64) return;
  int e = (int)(idx >> 6), c = (int)(idx & 63);
  int s = src[e], d = dst[e];
  float g = 1.f / (1.f + __expf(-(cd[s] - cd[d])));
  atomicAdd(&agg[(long)d * 64 + c], g * x[(long)s * 64 + c]);
}

__global__ void k_add2(const float* __restrict__ a, const float* __restrict__ b,
                       float* __restrict__ out, long count)
{
  long i = (long)blockIdx.x * blockDim.x + threadIdx.x;
  if (i < count) out[i] = a[i] + b[i];
}

__global__ void k_concat_h(const float* __restrict__ hci, const float* __restrict__ hsi,
                           const float* __restrict__ hns, float* __restrict__ h, int n)
{
  long idx = (long)blockIdx.x * blockDim.x + threadIdx.x;
  if (idx >= (long)n * 192) return;
  int node = (int)(idx / 192), c = (int)(idx % 192);
  float v;
  if (c < 64)       v = hci[(long)node * 64 + c];
  else if (c < 128) v = hsi[(long)node * 64 + c - 64];
  else              v = hns[(long)node * 64 + c - 128];
  h[idx] = v;
}

__global__ void k_softmax192(float* __restrict__ L, int n)
{
  int node = blockIdx.x, tid = threadIdx.x;     // blockDim = 256
  __shared__ float red[256];
  float v = (tid < 192) ? L[(long)node * 192 + tid] : -3.0e38f;
  red[tid] = v; __syncthreads();
  for (int s = 128; s > 0; s >>= 1) { if (tid < s) red[tid] = fmaxf(red[tid], red[tid + s]); __syncthreads(); }
  float mx = red[0]; __syncthreads();
  float ex = (tid < 192) ? __expf(v - mx) : 0.f;
  red[tid] = ex; __syncthreads();
  for (int s = 128; s > 0; s >>= 1) { if (tid < s) red[tid] += red[tid + s]; __syncthreads(); }
  float inv = 1.f / red[0];
  if (tid < 192) L[(long)node * 192 + tid] = ex * inv;
}

__global__ void k_mix(const float* __restrict__ A, const float* __restrict__ hci,
                      const float* __restrict__ hsi, const float* __restrict__ hns,
                      float* __restrict__ p, int n)
{
  long idx = (long)blockIdx.x * blockDim.x + threadIdx.x;
  if (idx >= (long)n * 64) return;
  int node = (int)(idx >> 6), c = (int)(idx & 63);
  const float* a = A + (long)node * 192;
  p[idx] = a[c] * hci[idx] + a[64 + c] * hsi[idx] + a[128 + c] * hns[idx];
}

__global__ void k_dot64(const float* __restrict__ X, const float* __restrict__ w,
                        const float* __restrict__ b, float* __restrict__ out, int n)
{
  int i = blockIdx.x * blockDim.x + threadIdx.x;
  if (i >= n) return;
  float acc = b[0];
  const float* xr = X + (long)i * 64;
  for (int c = 0; c < 64; ++c) acc += xr[c] * w[c];
  out[i] = acc;
}

__global__ void k_final(const float* __restrict__ y0, const float* __restrict__ y1,
                        const float* __restrict__ t, float* __restrict__ out, int n)
{
  int i = blockIdx.x * blockDim.x + threadIdx.x;
  if (i >= n) return;
  float tv = t[i], a = y0[i], b = y1[i];
  out[i]         = (1.f - tv) * a + tv * b;   // pred_y
  out[n + i]     = tv * a + (1.f - tv) * b;   // pred_y_cf
  out[2*n + i]   = a;                         // pred_y0
  out[3*n + i]   = b;                         // pred_y1
}

// ======================================================================
// Host launchers
// ======================================================================
template<typename TX>
static inline void launch_gemm(hipStream_t st, const TX* X, int ldx, const _Float16* Wp,
                               const float* bias, float* Out, int ldo,
                               int N, int K, int M, int act,
                               const TX* Xadd, const float* rowScale,
                               const float* addPre, const int* preIdx,
                               const float* addPost, const int* scatIdx)
{
  int tilesMG = (M + 63) / 64;
  long waves = (long)((N + 15) / 16) * tilesMG;
  long blocks = (waves + 7) / 8;
  wmma_gemm_kernel<TX><<<dim3((unsigned)blocks), dim3(256), 0, st>>>(
      X, ldx, Xadd, rowScale, Wp, bias, addPre, preIdx, addPost, Out, ldo, scatIdx,
      N, K, M, act, tilesMG);
}

static inline void gemmF(hipStream_t st, const float* X, int ldx, const _Float16* Wp,
                         const float* bias, float* Out, int ldo, int N, int K, int M, int act,
                         const float* Xadd = nullptr, const float* rowScale = nullptr,
                         const float* addPre = nullptr, const int* preIdx = nullptr,
                         const float* addPost = nullptr, const int* scatIdx = nullptr)
{
  launch_gemm<float>(st, X, ldx, Wp, bias, Out, ldo, N, K, M, act,
                     Xadd, rowScale, addPre, preIdx, addPost, scatIdx);
}

static inline void gemmH(hipStream_t st, const _Float16* X, int ldx, const _Float16* Wp,
                         const float* bias, float* Out, int ldo, int N, int K, int M, int act,
                         const float* addPre = nullptr, const int* preIdx = nullptr,
                         const int* scatIdx = nullptr)
{
  launch_gemm<_Float16>(st, X, ldx, Wp, bias, Out, ldo, N, K, M, act,
                        (const _Float16*)nullptr, (const float*)nullptr,
                        addPre, preIdx, (const float*)nullptr, scatIdx);
}

extern "C" void kernel_launch(void* const* d_in, const int* in_sizes, int n_in,
                              void* d_out, int out_size, void* d_ws, size_t ws_size,
                              hipStream_t stream)
{
  const int n = in_sizes[0] / 38;
  const int E = in_sizes[2] / 2;

  const float* Xd  = (const float*)d_in[0];   // discrete_x (n,38)
  const float* Xc  = (const float*)d_in[1];   // continous_x (n,3,16)
  const int*   src = (const int*)d_in[2];     // edge_index[0]
  const int*   dst = src + E;                 // edge_index[1]
  const float* EA  = (const float*)d_in[3];   // edge_attr (E,8)
  const float* CD  = (const float*)d_in[4];   // churn_date (n)
  const float* T   = (const float*)d_in[5];   // t (n,1)
  int pi = 6;
  const float* W_c   = (const float*)d_in[pi++]; const float* b_c   = (const float*)d_in[pi++];
  const float* W_g0  = (const float*)d_in[pi++]; const float* b_g0  = (const float*)d_in[pi++];
  const float* W_gcn1= (const float*)d_in[pi++]; const float* b_gcn1= (const float*)d_in[pi++];
  const float* W_gcn2= (const float*)d_in[pi++]; const float* b_gcn2= (const float*)d_in[pi++];
  const float* W_e   = (const float*)d_in[pi++]; const float* b_e   = (const float*)d_in[pi++];
  const float* W_r1  = (const float*)d_in[pi++]; const float* b_r1  = (const float*)d_in[pi++];
  const float* W_r2  = (const float*)d_in[pi++]; const float* b_r2  = (const float*)d_in[pi++];
  const float* w_x0  = (const float*)d_in[pi++]; const float* b_x0  = (const float*)d_in[pi++];
  const float* w_x1  = (const float*)d_in[pi++]; const float* b_x1  = (const float*)d_in[pi++];
  const float* W_f   = (const float*)d_in[pi++]; const float* b_f   = (const float*)d_in[pi++];
  const float* W_sg0 = (const float*)d_in[pi++]; const float* b_sg0 = (const float*)d_in[pi++];
  const float* W_n1  = (const float*)d_in[pi++]; const float* W_ee1 = (const float*)d_in[pi++];
  const float* W_rt1 = (const float*)d_in[pi++]; const float* b_el1 = (const float*)d_in[pi++];
  const float* W_n2  = (const float*)d_in[pi++]; const float* W_ee2 = (const float*)d_in[pi++];
  const float* W_rt2 = (const float*)d_in[pi++]; const float* b_el2 = (const float*)d_in[pi++];
  const float* W_sc0 = (const float*)d_in[pi++]; const float* b_sc0 = (const float*)d_in[pi++];
  const float* W_ga  = (const float*)d_in[pi++]; const float* W_gs  = (const float*)d_in[pi++];
  const float* b_gg  = (const float*)d_in[pi++];
  const float* W_a0  = (const float*)d_in[pi++]; const float* b_a0  = (const float*)d_in[pi++];
  const float* W_a1  = (const float*)d_in[pi++]; const float* b_a1  = (const float*)d_in[pi++];
  const float* W_y0h = (const float*)d_in[pi++]; const float* b_y0h = (const float*)d_in[pi++];
  const float* W_y0o = (const float*)d_in[pi++]; const float* b_y0o = (const float*)d_in[pi++];
  const float* W_y1h = (const float*)d_in[pi++]; const float* b_y1h = (const float*)d_in[pi++];
  const float* W_y1o = (const float*)d_in[pi++]; const float* b_y1o = (const float*)d_in[pi++];
  const float* W_T   = (const float*)d_in[pi++]; const float* b_T   = (const float*)d_in[pi++];

  // ---- workspace bump allocator ----
  char* wsb = (char*)d_ws;
  size_t off = 0;
  auto allocB = [&](size_t bytes) -> void* {
    void* p = (void*)(wsb + off);
    off += (bytes + 255) & ~(size_t)255;
    return p;
  };
  auto allocF = [&](long elems) -> float* { return (float*)allocB((size_t)elems * 4); };

  float* x      = allocF((long)n * 120);
  float* x_g    = allocF((long)n * 64);
  float* hg     = allocF((long)n * 64);
  float* agg    = allocF((long)n * 64);
  float* x_g0   = allocF((long)n * 64);
  float* x_g1   = allocF((long)n * 64);
  float* h1     = allocF((long)n * 120);
  float* x_deep = allocF((long)n * 120);
  float* xl     = allocF((long)n * 120);
  float* xl2    = allocF((long)n * 120);
  float* cnt    = allocF(n);
  float* deg    = allocF(n);
  float* inv_deg  = allocF(n);
  float* inv_degm = allocF(n);
  float* x_si   = allocF((long)n * 64);
  float* A1     = allocF((long)n * 64);
  float* A2     = allocF((long)n * 64);
  float* xn     = allocF((long)n * 64);
  float* xrt    = allocF((long)n * 64);
  float* x_si0  = allocF((long)n * 64);
  float* x_si1  = allocF((long)n * 64);
  float* x_ns   = allocF((long)n * 64);
  float* x_ns0  = allocF((long)n * 64);
  float* x_ns1  = allocF((long)n * 64);
  float* G      = allocF((long)n * 64);
  float* h_ns   = allocF((long)n * 64);
  float* hbuf   = allocF((long)n * 192);
  float* L0     = allocF((long)n * 192);
  float* L1     = allocF((long)n * 192);
  float* p0     = allocF((long)n * 64);
  float* p1     = allocF((long)n * 64);
  float* hy0    = allocF((long)n * 64);
  float* hy1    = allocF((long)n * 64);
  float* y0     = allocF(n);
  float* y1     = allocF(n);
  _Float16* eH  = (_Float16*)allocB((size_t)E * 64 * 2);

  auto EWB = [](long c) { return (unsigned)((c + 255) / 256); };

  // ---- pack all GEMM weights to f16 WMMA-B layout (one-time, tiny) ----
  auto packW = [&](const float* W, int ldw, int K, int M) -> _Float16* {
    int nCh = (K + 31) / 32;
    long total = (long)nCh * M * 32;
    _Float16* p = (_Float16*)allocB((size_t)total * 2);
    k_packW<<<EWB(total), 256, 0, stream>>>(W, ldw, K, M, p, total);
    return p;
  };
  _Float16* P_g0   = packW(W_g0,  64,  56,  64);
  _Float16* P_gcn1 = packW(W_gcn1,64,  64,  64);
  _Float16* P_gcn2 = packW(W_gcn2,64,  64,  64);
  _Float16* P_r1   = packW(W_r1, 120, 120, 120);
  _Float16* P_r2   = packW(W_r2, 120, 120, 120);
  _Float16* P_f    = packW(W_f,   64, 120,  64);
  _Float16* P_sg0  = packW(W_sg0, 64, 120,  64);
  _Float16* P_eT   = packW(W_e,            64, 64, 64);   // rows 0:64
  _Float16* P_eM   = packW(W_e + 64 * 64,  64, 64, 64);   // rows 64:128
  _Float16* P_n1   = packW(W_n1,  64,  64,  64);
  _Float16* P_rt1  = packW(W_rt1, 64,  64,  64);
  _Float16* P_ee1  = packW(W_ee1, 64,  64,  64);
  _Float16* P_n2   = packW(W_n2,  64,  64,  64);
  _Float16* P_rt2  = packW(W_rt2, 64,  64,  64);
  _Float16* P_ee2  = packW(W_ee2, 64,  64,  64);
  _Float16* P_sc0  = packW(W_sc0, 64, 120,  64);
  _Float16* P_ga   = packW(W_ga,  64,  64,  64);
  _Float16* P_gs   = packW(W_gs,  64,  64,  64);
  _Float16* P_a0   = packW(W_a0, 192, 192, 192);
  _Float16* P_a1   = packW(W_a1, 192, 192, 192);
  _Float16* P_y0h  = packW(W_y0h, 64,  64,  64);
  _Float16* P_y1h  = packW(W_y1h, 64,  64,  64);

  float* outF = (float*)d_out;
  float* h_ci = outF + (long)5 * n;                 // (n,64)
  float* h_si = outF + (long)5 * n + (long)64 * n;  // (n,64)
  const size_t aggB = (size_t)n * 64 * 4;

  // ---- degrees ----
  hipMemsetAsync(cnt, 0, (size_t)n * 4, stream);
  k_count<<<EWB(E), 256, 0, stream>>>(dst, cnt, E);
  k_deg<<<EWB(n), 256, 0, stream>>>(cnt, deg, inv_deg, inv_degm, n);

  // ---- embedding: x[:, :56] = [x_d | relu(conv1d(cont))] ----
  k_embed<<<EWB(n), 256, 0, stream>>>(Xd, Xc, W_c, b_c, x, n);

  // ---- x_g = relu(x_dc @ W_g0 + b) ----
  gemmF(stream, x, 120, P_g0, b_g0, x_g, 64, n, 56, 64, 1);

  // ---- GCN layer 1 ----
  gemmF(stream, x_g, 64, P_gcn1, nullptr, hg, 64, n, 64, 64, 0);
  hipMemsetAsync(agg, 0, aggB, stream);
  k_gcn_scatter<<<EWB((long)E * 64), 256, 0, stream>>>(src, dst, hg, deg, agg, E);
  k_gcn_combine<<<EWB((long)n * 64), 256, 0, stream>>>(agg, hg, inv_deg, b_gcn1, x_g0, n);
  // ---- GCN layer 2 ----
  gemmF(stream, x_g0, 64, P_gcn2, nullptr, hg, 64, n, 64, 64, 0);
  hipMemsetAsync(agg, 0, aggB, stream);
  k_gcn_scatter<<<EWB((long)E * 64), 256, 0, stream>>>(src, dst, hg, deg, agg, E);
  k_gcn_combine<<<EWB((long)n * 64), 256, 0, stream>>>(agg, hg, inv_deg, b_gcn2, x_g1, n);
  // ---- x[:, 56:120] = x_g0 + x_g1 ----
  k_xsum<<<EWB((long)n * 64), 256, 0, stream>>>(x_g0, x_g1, x, n);

  // ---- deep branch ----
  gemmF(stream, x, 120, P_r1, b_r1, h1, 120, n, 120, 120, 1,
        nullptr, nullptr, nullptr, nullptr, x);           // relu(..)+x
  gemmF(stream, h1, 120, P_r2, b_r2, x_deep, 120, n, 120, 120, 1,
        nullptr, nullptr, nullptr, nullptr, h1);          // relu(..)+h1
  // ---- cross branch (xl starts at x) ----
  k_cross<<<n, 128, 0, stream>>>(x, x, w_x0, b_x0, xl, n);
  k_cross<<<n, 128, 0, stream>>>(x, xl, w_x1, b_x1, xl2, n);
  // ---- h_ci = relu((x_deep + xl2) @ W_f + b_f)  -> d_out ----
  gemmF(stream, x_deep, 120, P_f, b_f, h_ci, 64, n, 120, 64, 1, xl2);

  // ---- social-influence branch ----
  gemmF(stream, x, 120, P_sg0, b_sg0, x_si, 64, n, 120, 64, 1);
  // e = relu(A1[src] + A2[dst] + ea@W_e[128:] + b_e), stored f16
  gemmF(stream, x_si, 64, P_eT, nullptr, A1, 64, n, 64, 64, 0);
  gemmF(stream, x_si, 64, P_eM, nullptr, A2, 64, n, 64, 64, 0);
  k_edge_e<<<EWB((long)E * 64), 256, 0, stream>>>(src, dst, A1, A2, EA,
                                                  W_e + 128 * 64, b_e, eH, E);
  // EL-conv layer 1: agg = seg_sum(e@W_ee1 + (x_si@W_n1)[src], dst) fused scatter
  gemmF(stream, x_si, 64, P_n1,  nullptr, xn,  64, n, 64, 64, 0);
  gemmF(stream, x_si, 64, P_rt1, nullptr, xrt, 64, n, 64, 64, 0);
  hipMemsetAsync(agg, 0, aggB, stream);
  gemmH(stream, eH, 64, P_ee1, nullptr, agg, 64, E, 64, 64, 0, xn, src, dst);
  k_si_combine<<<EWB((long)n * 64), 256, 0, stream>>>(agg, inv_degm, xrt, b_el1, x_si0, n);
  // EL-conv layer 2
  gemmF(stream, x_si0, 64, P_n2,  nullptr, xn,  64, n, 64, 64, 0);
  gemmF(stream, x_si0, 64, P_rt2, nullptr, xrt, 64, n, 64, 64, 0);
  hipMemsetAsync(agg, 0, aggB, stream);
  gemmH(stream, eH, 64, P_ee2, nullptr, agg, 64, E, 64, 64, 0, xn, src, dst);
  k_si_combine<<<EWB((long)n * 64), 256, 0, stream>>>(agg, inv_degm, xrt, b_el2, x_si1, n);
  k_add2<<<EWB((long)n * 64), 256, 0, stream>>>(x_si0, x_si1, h_si, (long)n * 64);

  // ---- gated branch ----
  gemmF(stream, x, 120, P_sc0, b_sc0, x_ns, 64, n, 120, 64, 1);
  hipMemsetAsync(agg, 0, aggB, stream);
  k_gate_scatter<<<EWB((long)E * 64), 256, 0, stream>>>(src, dst, CD, x_ns, agg, E);
  gemmF(stream, agg, 64, P_ga, nullptr, G, 64, n, 64, 64, 0,
        nullptr, inv_degm);                                 // (agg/degm) @ W_ga
  gemmF(stream, x_ns, 64, P_gs, b_gg, x_ns0, 64, n, 64, 64, 1,
        nullptr, nullptr, G, nullptr);                      // relu(G + x@Ws + b)
  hipMemsetAsync(agg, 0, aggB, stream);
  k_gate_scatter<<<EWB((long)E * 64), 256, 0, stream>>>(src, dst, CD, x_ns0, agg, E);
  gemmF(stream, agg, 64, P_ga, nullptr, G, 64, n, 64, 64, 0,
        nullptr, inv_degm);
  gemmF(stream, x_ns0, 64, P_gs, b_gg, x_ns1, 64, n, 64, 64, 1,
        nullptr, nullptr, G, nullptr);
  k_add2<<<EWB((long)n * 64), 256, 0, stream>>>(x_ns0, x_ns1, h_ns, (long)n * 64);

  // ---- pred_T ----
  k_dot64<<<EWB(n), 256, 0, stream>>>(h_si, W_T, b_T, outF + (long)4 * n, n);

  // ---- attention heads + mixing ----
  k_concat_h<<<EWB((long)n * 192), 256, 0, stream>>>(h_ci, h_si, h_ns, hbuf, n);
  gemmF(stream, hbuf, 192, P_a0, b_a0, L0, 192, n, 192, 192, 0);
  gemmF(stream, hbuf, 192, P_a1, b_a1, L1, 192, n, 192, 192, 0);
  k_softmax192<<<n, 256, 0, stream>>>(L0, n);
  k_softmax192<<<n, 256, 0, stream>>>(L1, n);
  k_mix<<<EWB((long)n * 64), 256, 0, stream>>>(L0, h_ci, h_si, h_ns, p0, n);
  k_mix<<<EWB((long)n * 64), 256, 0, stream>>>(L1, h_ci, h_si, h_ns, p1, n);

  // ---- output MLPs ----
  gemmF(stream, p0, 64, P_y0h, b_y0h, hy0, 64, n, 64, 64, 1);
  gemmF(stream, p1, 64, P_y1h, b_y1h, hy1, 64, n, 64, 64, 1);
  k_dot64<<<EWB(n), 256, 0, stream>>>(hy0, W_y0o, b_y0o, y0, n);
  k_dot64<<<EWB(n), 256, 0, stream>>>(hy1, W_y1o, b_y1o, y1, n);
  k_final<<<EWB(n), 256, 0, stream>>>(y0, y1, T, outF, n);
}
